// Linear4bit_77309411567
// MI455X (gfx1250) — 8-bit matrix operand order repaired and checked by reading, compiled, not run on hardware
//
#include <hip/hip_runtime.h>

typedef __attribute__((ext_vector_type(8))) int          v8i;
typedef __attribute__((ext_vector_type(4))) int          v4i;
typedef __attribute__((ext_vector_type(2))) unsigned int v2u;

namespace {
constexpr int MDIM = 4096;
constexpr int KDIM = 4096;
constexpr int NDIM = 11008;
constexpr int KP   = KDIM / 2;
constexpr int BM = 128, BN = 128, BK = 64;
constexpr int NSTEP = KDIM / BK;
constexpr int KPB = BK / 2;
constexpr unsigned STEP_BYTES = KPB * 4;
constexpr unsigned CHUNK_BYTES = 32u * KP * 4u;
constexpr int A_STRIDE = 72;
constexpr int B_STRIDE = 80;
constexpr int A_BYTES  = BM * A_STRIDE;
constexpr int BUF_BYTES = A_BYTES + BN * B_STRIDE;
}

__device__ __forceinline__ unsigned int unpack2(unsigned int v0, unsigned int v1) {
  unsigned int a = v0 | (v0 << 4);
  unsigned int c = v1 | (v1 << 4);
  unsigned int u = (a | (c << 16)) & 0x0F0F0F0Fu;
  return (u + 0x78787878u) ^ 0x78787878u;
}

__global__ __launch_bounds__(256)
void linear4bit_i4_wmma(const int* __restrict__ xq,
                        const _Float16* __restrict__ sx,
                        const int* __restrict__ wq,
                        const _Float16* __restrict__ sw,
                        _Float16* __restrict__ out) {
  __shared__ __align__(16) unsigned char smem[2 * BUF_BYTES];

  const int tid  = threadIdx.x;
  const int wave = tid >> 5;
  const int lane = tid & 31;
  const int m0 = blockIdx.y * BM;
  const int n0 = blockIdx.x * BN;

  const unsigned srow0 = (unsigned)tid >> 3;
  const unsigned scol0 = ((unsigned)tid & 7) * 4;
  const unsigned char* xqb = (const unsigned char*)xq;
  const unsigned char* wqb = (const unsigned char*)wq;
  const unsigned aOff0 = (((unsigned)m0 + srow0) * (unsigned)KP + scol0) * 4u;
  const unsigned bOff0 = (((unsigned)n0 + srow0) * (unsigned)KP + scol0) * 4u;
  const unsigned sA = srow0 * A_STRIDE + scol0 * 2;
  const unsigned sB = srow0 * B_STRIDE + scol0 * 2;

  v4i pa[4], pb[4];
  auto load_step = [&](unsigned aO, unsigned bO) {
#pragma unroll
    for (int j = 0; j < 4; ++j) {
      pa[j] = *(const v4i*)(xqb + aO + (unsigned)j * CHUNK_BYTES);
      pb[j] = *(const v4i*)(wqb + bO + (unsigned)j * CHUNK_BYTES);
    }
  };
  auto store_step = [&](int buf) {
    unsigned char* bufA = smem + buf * BUF_BYTES;
    unsigned char* bufB = bufA + A_BYTES;
#pragma unroll
    for (int j = 0; j < 4; ++j) {
      v2u ua = { unpack2((unsigned)pa[j].x, (unsigned)pa[j].y),
                 unpack2((unsigned)pa[j].z, (unsigned)pa[j].w) };
      v2u ub = { unpack2((unsigned)pb[j].x, (unsigned)pb[j].y),
                 unpack2((unsigned)pb[j].z, (unsigned)pb[j].w) };
      *(v2u*)(bufA + sA + j * (32 * A_STRIDE)) = ua;
      *(v2u*)(bufB + sB + j * (32 * B_STRIDE)) = ub;
    }
  };

  v8i acc[8] = {};

  const int aoff = (lane >> 4) << 4;
  const int boff = (lane >> 4) << 4;
  const int arow = wave * 16 + (lane & 15);

  load_step(aOff0, bOff0);
  store_step(0);
  load_step(aOff0 + STEP_BYTES, bOff0 + STEP_BYTES);
  unsigned aNext = aOff0 + 2 * STEP_BYTES;
  unsigned bNext = bOff0 + 2 * STEP_BYTES;

  int buf = 0;
  for (int step = 0; step < NSTEP; ++step) {
    __syncthreads();

    const unsigned char* bufA = smem + buf * BUF_BYTES;
    const unsigned char* bufB = bufA + A_BYTES;

    const unsigned char* rA = bufA + arow * A_STRIDE + aoff;
    v2u a0 = *(const v2u*)(rA);
    v2u a1 = *(const v2u*)(rA + 8);
    v2u a2 = *(const v2u*)(rA + 32);
    v2u a3 = *(const v2u*)(rA + 40);
    v8i a = { (int)a0.x, (int)a0.y, (int)a1.x, (int)a1.y,
              (int)a2.x, (int)a2.y, (int)a3.x, (int)a3.y };

    auto loadB = [&](int j) -> v8i {
      const unsigned char* rB = bufB + (j * 16 + (lane & 15)) * B_STRIDE + boff;
      v4i b0 = *(const v4i*)(rB);
      v4i b1 = *(const v4i*)(rB + 32);
      return v8i{ b0.x, b0.y, b0.z, b0.w, b1.x, b1.y, b1.z, b1.w };
    };
    v8i bfr[2];
    bfr[0] = loadB(0);
#pragma unroll
    for (int j = 0; j < 8; ++j) {
      if (j + 1 < 8) bfr[(j + 1) & 1] = loadB(j + 1);
      acc[j] = __builtin_amdgcn_wmma_i32_16x16x64_iu8(
           true, a,  true, bfr[j & 1], acc[j],
           false,  false);
    }

#if __has_builtin(__builtin_amdgcn_sched_group_barrier)
    __builtin_amdgcn_sched_group_barrier(0x100, 6, 0);
#pragma unroll
    for (int g = 0; g < 6; ++g) {
      __builtin_amdgcn_sched_group_barrier(0x008, 1, 0);
      __builtin_amdgcn_sched_group_barrier(0x100, 2, 0);
    }
    __builtin_amdgcn_sched_group_barrier(0x008, 2, 0);
#endif

    if (step + 1 < NSTEP) store_step(buf ^ 1);
    if (step + 2 < NSTEP) {
      load_step(aNext, bNext);
      aNext += STEP_BYTES;
      bNext += STEP_BYTES;
    }
    buf ^= 1;
  }

  const int baseRow = m0 + wave * 16 + ((lane >> 4) << 3);
  const int col0    = n0 + (lane & 15);
  float swv[8];
#pragma unroll
  for (int j = 0; j < 8; ++j) swv[j] = (float)sw[col0 + j * 16];
#pragma unroll
  for (int r = 0; r < 8; ++r) {
    const int row = baseRow + r;
    const float sxv = (float)sx[row];
    _Float16* orow = out + (size_t)row * NDIM;
#pragma unroll
    for (int j = 0; j < 8; ++j) {
      orow[col0 + j * 16] = (_Float16)((float)acc[j][r] * sxv * swv[j]);
    }
  }
}

extern "C" void kernel_launch(void* const* d_in, const int* in_sizes, int n_in,
                              void* d_out, int out_size, void* d_ws, size_t ws_size,
                              hipStream_t stream) {
  const int*      xq = (const int*)d_in[0];
  const _Float16* sx = (const _Float16*)d_in[1];
  const int*      wq = (const int*)d_in[2];
  const _Float16* sw = (const _Float16*)d_in[3];
  _Float16* out = (_Float16*)d_out;

  dim3 grid(NDIM / BN, MDIM / BM);
  dim3 block(256);
  hipLaunchKernelGGL(linear4bit_i4_wmma, grid, block, 0, stream,
                     xq, sx, wq, sw, out);
}
